// stableTransMIL_8143257993461
// MI455X (gfx1250) — matrix operand order repaired and checked by reading, not compiled, not run on hardware
//
#include <hip/hip_runtime.h>

// ---------------- types for CDNA5 WMMA ----------------
typedef __bf16 bf16_t;
typedef __attribute__((ext_vector_type(16))) __bf16 v16bf;
typedef __attribute__((ext_vector_type(8)))  __bf16 bf16x8;
typedef __attribute__((ext_vector_type(8)))  float  v8f;

union FragU { v16bf v; bf16x8 h[2]; };

// ---------------- problem constants ----------------
#define DM    1024      // model dim
#define NSEQ  9999      // raw tokens
#define NP    10240     // padded sequence (multiple of 512 landmarks)
#define PADR  239       // zero-pad rows at front (512 - 10001%512)
#define HEADS 8
#define DH    128
#define MLM   512       // landmarks
#define LFAC  20        // NP / MLM
#define GH    100       // ppeg grid 100x100

#define BM 128
#define BN 128
#define BK 32
#define LDT 40                      // padded LDS row stride (elements): 64B row + 16B pad
#define TILEB (BM * LDT * 2)        // one staged tile, bytes (10240)
#define GEMM_SMEM (4 * TILEB)       // double-buffered As+Bs, bytes (40960)

__device__ __forceinline__ float bf2f(bf16_t b) { return (float)b; }
__device__ __forceinline__ bf16_t f2bf(float f) { return (bf16_t)f; }

// ---------------- gfx1250 Tensor Data Mover path (guarded) ----------------
#if defined(__HIP_DEVICE_COMPILE__) && defined(__gfx1250__) && \
    __has_builtin(__builtin_amdgcn_tensor_load_to_lds) &&      \
    __has_builtin(__builtin_amdgcn_s_wait_tensorcnt)
#define HAVE_TDM 1
#else
#define HAVE_TDM 0
#endif

#if HAVE_TDM
typedef __attribute__((ext_vector_type(4))) unsigned int u32x4;
typedef __attribute__((ext_vector_type(4))) int i32x4;
typedef __attribute__((ext_vector_type(8))) int i32x8;

#if __has_include(<hip/amd_detail/amd_gfx1250_TDM.h>)
// amdgpu-toolchain (clang-23 / therock-10.0): 6-arg builtin
#define TDM_ISSUE(g0, g1) \
  __builtin_amdgcn_tensor_load_to_lds((g0), (g1), (i32x4)0, (i32x4)0, (i32x8)0, 0)
#else
// ROCm 7.2 (clang-22): 5-arg builtin
#define TDM_ISSUE(g0, g1) \
  __builtin_amdgcn_tensor_load_to_lds((g0), (g1), (i32x4)0, (i32x4)0, 0)
#endif

#define TDM_TDIM1 (1u << 20)   // generous tensor_dim1 (rows); tiles always in-bounds

// DMA a 128x32 bf16 tile (global row stride = strideElems) into LDS at byte
// offset ldsOff, inserting 4 DWORDs of pad after every 16 DWORDs so the LDS
// pitch is LDT=40 elements (matches the WMMA fragment loader).
__device__ __forceinline__ void tdm_load_tile(unsigned ldsOff, const bf16_t* g, int strideElems)
{
  unsigned long long ga = (unsigned long long)(size_t)g;
  u32x4 g0;
  g0[0] = 1u;                                                  // count=1 user D#
  g0[1] = ldsOff;                                              // lds_addr (bytes)
  g0[2] = (unsigned)ga;                                        // global_addr[31:0]
  g0[3] = (unsigned)((ga >> 32) & 0x01FFFFFFu) | (2u << 30);   // [56:32] | type=2
  i32x8 g1;
  g1[0] = (int)((1u << 16) |                                   // data_size = 2B
                (1u << 20) |                                   // pad_enable
                (3u << 22) |                                   // pad_interval = 16 DW
                (3u << 25));                                   // pad_amount   = 4 DW
  g1[1] = (int)(((unsigned)strideElems & 0xFFFFu) << 16);      // tensor_dim0 lo16
  g1[2] = (int)((((unsigned)strideElems >> 16) & 0xFFFFu) |
                ((TDM_TDIM1 & 0xFFFFu) << 16));                // dim0 hi16 | dim1 lo16
  g1[3] = (int)(((TDM_TDIM1 >> 16) & 0xFFFFu) | (32u << 16));  // dim1 hi16 | tile_dim0=32
  g1[4] = (int)128u;                                           // tile_dim1=128, tile_dim2=0
  g1[5] = (int)(unsigned)strideElems;                          // tensor_dim0_stride lo32
  g1[6] = 0;                                                   // stride hi | dim1_stride lo
  g1[7] = 0;
  TDM_ISSUE(g0, g1);
}
#endif // HAVE_TDM

// ---------------- GEMM epilogue descriptor ----------------
struct EpArgs {
  int   mode;        // 0=f32  1=bf16  2=qkv-split  3=bias+resid f32 (zero below row)
  float alpha;
  float*       outF;
  bf16_t*      outB;
  const float* bias;
  const float* resid;
  int   ldo;
  long long strideO;
  int   zeroBelow;
  bf16_t* qp; bf16_t* kp; bf16_t* vp;
  float qscale;
  long long headStride;
};

// ---------------- batched bf16 WMMA GEMM: C = alpha * A(MxK) * B(KxN) ----------------
// A row-major [M][lda].  B: transB==0 -> row-major [K][ldb]; transB==1 -> stored [N][ldb] (B^T source).
// Double-buffered LDS tiles: Tensor Data Mover DMA for tile k+1 overlaps WMMA on tile k.
__global__ __launch_bounds__(256) void gemm_bf16_kernel(
    const bf16_t* __restrict__ A, const bf16_t* __restrict__ B,
    int M, int N, int K, int lda, int ldb,
    long long sA, long long sB, int transB, EpArgs ep)
{
  extern __shared__ __align__(16) char smemRaw[];

  const int tid  = threadIdx.x;
  const int lane = tid & 31;
  const int wave = tid >> 5;
  const int wm   = wave & 3;   // 4 row groups of 32
  const int wn   = wave >> 2;  // 2 col groups of 64
  const int bm0  = blockIdx.y * BM;
  const int bn0  = blockIdx.x * BN;
  const int z    = blockIdx.z;

  A += (long long)z * sA;
  B += (long long)z * sB;

#if HAVE_TDM
  const unsigned ldsBase = (unsigned)__builtin_amdgcn_groupstaticsize();
#endif

  v8f acc[2][4];
  #pragma unroll
  for (int i = 0; i < 2; ++i)
    #pragma unroll
    for (int j = 0; j < 4; ++j)
      #pragma unroll
      for (int e = 0; e < 8; ++e) acc[i][j][e] = 0.0f;

  const int lr = lane & 15;
  const int kh = lane >> 4;

  // stage tile at K-offset kk into LDS buffer `buf` (issue only; no waits here)
  auto stage = [&](int buf, int kk) {
    bf16_t* Asb = (bf16_t*)(smemRaw + (size_t)buf * 2 * TILEB);
    bf16_t* Bsb = (bf16_t*)(smemRaw + (size_t)buf * 2 * TILEB + TILEB);
#if HAVE_TDM
    if (tid < 32) {                                  // wave 0 issues DMAs
      tdm_load_tile(ldsBase + (unsigned)(buf * 2 * TILEB),
                    &A[(long long)bm0 * lda + kk], lda);
      if (transB)
        tdm_load_tile(ldsBase + (unsigned)(buf * 2 * TILEB + TILEB),
                      &B[(long long)bn0 * ldb + kk], ldb);
    }
    if (!transB) {
      // on-the-fly transpose: B[k][n] -> Bs[n][k] (TDM cannot transpose)
      #pragma unroll
      for (int c = 0; c < 2; ++c) {
        int chunk = tid * 2 + c;          // 32 k rows x 16 n-chunks
        int kkk   = chunk >> 4;           // 0..31
        int nb    = (chunk & 15) * 8;     // 0..120
        bf16x8 vdat = *(const bf16x8*)&B[(long long)(kk + kkk) * ldb + bn0 + nb];
        #pragma unroll
        for (int e = 0; e < 8; ++e) Bsb[(nb + e) * LDT + kkk] = vdat[e];
      }
    }
    (void)Asb;
#else
    #pragma unroll
    for (int c = 0; c < 2; ++c) {
      int chunk = tid * 2 + c;           // 0..511
      int row   = chunk >> 2;            // 0..127
      int col   = (chunk & 3) * 8;       // 0,8,16,24
      *(bf16x8*)&Asb[row * LDT + col] =
          *(const bf16x8*)&A[(long long)(bm0 + row) * lda + kk + col];
    }
    if (transB) {
      #pragma unroll
      for (int c = 0; c < 2; ++c) {
        int chunk = tid * 2 + c;
        int row   = chunk >> 2;
        int col   = (chunk & 3) * 8;
        *(bf16x8*)&Bsb[row * LDT + col] =
            *(const bf16x8*)&B[(long long)(bn0 + row) * ldb + kk + col];
      }
    } else {
      #pragma unroll
      for (int c = 0; c < 2; ++c) {
        int chunk = tid * 2 + c;
        int kkk   = chunk >> 4;
        int nb    = (chunk & 15) * 8;
        bf16x8 vdat = *(const bf16x8*)&B[(long long)(kk + kkk) * ldb + bn0 + nb];
        #pragma unroll
        for (int e = 0; e < 8; ++e) Bsb[(nb + e) * LDT + kkk] = vdat[e];
      }
    }
#endif
  };

  // prologue: fill buffer 0 and publish it
  stage(0, 0);
#if HAVE_TDM
  if (tid < 32) __builtin_amdgcn_s_wait_tensorcnt(0);
#endif
  __syncthreads();

  int cur = 0;
  for (int k0 = 0; k0 < K; k0 += BK) {
    const int  nxt  = cur ^ 1;
    const bool more = (k0 + BK) < K;
    if (more) stage(nxt, k0 + BK);     // async: DMA/loads overlap the WMMAs below

    const bf16_t* Asb = (const bf16_t*)(smemRaw + (size_t)cur * 2 * TILEB);
    const bf16_t* Bsb = (const bf16_t*)(smemRaw + (size_t)cur * 2 * TILEB + TILEB);

    // ---- fragments per ISA 16-bit layouts ----
    FragU aF[2], bF[4];
    #pragma unroll
    for (int i = 0; i < 2; ++i) {
      int r = wm * 32 + i * 16 + lr;
      aF[i].h[0] = *(const bf16x8*)&Asb[r * LDT + kh * 8];       // K 0..7 / 8..15
      aF[i].h[1] = *(const bf16x8*)&Asb[r * LDT + 16 + kh * 8];  // K 16..23 / 24..31
    }
    #pragma unroll
    for (int j = 0; j < 4; ++j) {
      int r = wn * 64 + j * 16 + lr;
      bF[j].h[0] = *(const bf16x8*)&Bsb[r * LDT + kh * 8];       // K 0..7 / 8..15
      bF[j].h[1] = *(const bf16x8*)&Bsb[r * LDT + 16 + kh * 8];  // K 16..23 / 24..31
    }
    #pragma unroll
    for (int i = 0; i < 2; ++i)
      #pragma unroll
      for (int j = 0; j < 4; ++j)
        acc[i][j] = __builtin_amdgcn_wmma_f32_16x16x32_bf16(
            false, aF[i].v, false, bF[j].v, (short)0, acc[i][j], false, false);

#if HAVE_TDM
    if (more && tid < 32) __builtin_amdgcn_s_wait_tensorcnt(0);  // next tile DMA done
#endif
    __syncthreads();                   // publish next buffer; reads of cur complete
    cur = nxt;
  }

  // ---- epilogue ----
  #pragma unroll
  for (int i = 0; i < 2; ++i) {
    const int mBase = bm0 + wm * 32 + i * 16 + ((lane >> 4) << 3);
    #pragma unroll
    for (int j = 0; j < 4; ++j) {
      const int n = bn0 + wn * 64 + j * 16 + lr;
      #pragma unroll
      for (int e = 0; e < 8; ++e) {
        const int m = mBase + e;
        float val = acc[i][j][e] * ep.alpha;
        if (ep.mode == 0) {
          ep.outF[(long long)z * ep.strideO + (long long)m * ep.ldo + n] = val;
        } else if (ep.mode == 1) {
          ep.outB[(long long)z * ep.strideO + (long long)m * ep.ldo + n] = f2bf(val);
        } else if (ep.mode == 2) {
          int sect = n >> 10, head = (n >> 7) & 7, dd = n & 127;
          bf16_t* p = sect == 0 ? ep.qp : (sect == 1 ? ep.kp : ep.vp);
          float sc = sect == 0 ? ep.qscale : 1.0f;
          p[(long long)head * ep.headStride + (long long)m * DH + dd] = f2bf(val * sc);
        } else {
          float r = (m < ep.zeroBelow)
                        ? 0.0f
                        : (val + ep.bias[n] + ep.resid[(long long)m * ep.ldo + n]);
          ep.outF[(long long)m * ep.ldo + n] = r;
        }
      }
    }
  }
}

// ---------------- helpers ----------------
__device__ __forceinline__ float blockReduceSum(float v, float* red) {
  int tid = threadIdx.x;
  red[tid] = v; __syncthreads();
  for (int s = blockDim.x >> 1; s > 0; s >>= 1) {
    if (tid < s) red[tid] += red[tid + s];
    __syncthreads();
  }
  float r = red[0]; __syncthreads();
  return r;
}
__device__ __forceinline__ void atomicMaxPos(float* addr, float v) {
  atomicMax((int*)addr, __float_as_int(v));
}

// ---------------- build padded input sequence ----------------
__global__ __launch_bounds__(256) void build_kernel(
    const float* __restrict__ h_, const float* __restrict__ wgt,
    const float* __restrict__ cls, float* __restrict__ X)
{
  long long idx = (long long)blockIdx.x * 256 + threadIdx.x; // NP*DM
  int row = (int)(idx >> 10), c = (int)(idx & 1023);
  float v = 0.0f;
  if (row >= PADR) {
    int t = row - PADR;                // 0..10000
    if (t == 0) v = cls[c];
    else {
      int src = (t <= NSEQ) ? (t - 1) : 0;   // wrap row for square grid
      v = h_[(long long)src * DM + c] * wgt[src];
    }
  }
  X[idx] = v;
}

// ---------------- layernorm (in place) + bf16 copy; zero pad rows ----------------
__global__ __launch_bounds__(256) void ln_kernel(
    float* __restrict__ X, bf16_t* __restrict__ X16,
    const float* __restrict__ g, const float* __restrict__ b)
{
  __shared__ float red[256];
  int row = blockIdx.x, tid = threadIdx.x;
  float* xr = X + (long long)row * DM;
  bf16_t* xo = X16 + (long long)row * DM;
  if (row < PADR) {
    for (int c = tid; c < DM; c += 256) { xr[c] = 0.0f; xo[c] = f2bf(0.0f); }
    return;
  }
  float s = 0.0f;
  for (int c = tid; c < DM; c += 256) s += xr[c];
  float mu = blockReduceSum(s, red) * (1.0f / DM);
  float v = 0.0f;
  for (int c = tid; c < DM; c += 256) { float d = xr[c] - mu; v += d * d; }
  float var = blockReduceSum(v, red) * (1.0f / DM);
  float rs = rsqrtf(var + 1e-5f);
  for (int c = tid; c < DM; c += 256) {
    float val = (xr[c] - mu) * rs * g[c] + b[c];
    xr[c] = val; xo[c] = f2bf(val);
  }
}

// ---------------- landmark means over 20 consecutive tokens ----------------
__global__ __launch_bounds__(128) void landmark_kernel(
    const bf16_t* __restrict__ Q, const bf16_t* __restrict__ K,
    bf16_t* __restrict__ QL, bf16_t* __restrict__ KL)
{
  int h = blockIdx.x >> 9, i = blockIdx.x & 511, d = threadIdx.x;
  long long base = (long long)h * NP * DH + (long long)(i * LFAC) * DH + d;
  float sq = 0.0f, sk = 0.0f;
  #pragma unroll
  for (int j = 0; j < LFAC; ++j) {
    sq += bf2f(Q[base + (long long)j * DH]);
    sk += bf2f(K[base + (long long)j * DH]);
  }
  long long o = (long long)h * MLM * DH + (long long)i * DH + d;
  QL[o] = f2bf(sq * (1.0f / LFAC));
  KL[o] = f2bf(sk * (1.0f / LFAC));
}

// ---------------- row softmax: f32 in -> optional f32 out + bf16 out ----------------
__global__ __launch_bounds__(256) void softmax_kernel(
    const float* __restrict__ in, float* __restrict__ outF,
    bf16_t* __restrict__ outB, int cols)
{
  __shared__ float red[256];
  long long base = (long long)blockIdx.x * cols;
  int tid = threadIdx.x;
  float m = -3.4e38f;
  for (int c = tid; c < cols; c += 256) m = fmaxf(m, in[base + c]);
  red[tid] = m; __syncthreads();
  for (int s = 128; s > 0; s >>= 1) { if (tid < s) red[tid] = fmaxf(red[tid], red[tid + s]); __syncthreads(); }
  m = red[0]; __syncthreads();
  float s = 0.0f;
  for (int c = tid; c < cols; c += 256) s += __expf(in[base + c] - m);
  float inv = 1.0f / blockReduceSum(s, red);
  for (int c = tid; c < cols; c += 256) {
    float v = __expf(in[base + c] - m) * inv;
    if (outF) outF[base + c] = v;
    outB[base + c] = f2bf(v);
  }
}

// ---------------- pinv scale reductions ----------------
__global__ void init_scal_kernel(float* s1, float* s2) { s1[0] = 0.0f; s2[0] = 0.0f; }

__global__ __launch_bounds__(128) void rowmax_kernel(const float* __restrict__ a, float* s1)
{
  __shared__ float red[128];
  long long base = (long long)blockIdx.x * MLM;
  int tid = threadIdx.x;
  float s = 0.0f;
  for (int c = tid; c < MLM; c += 128) s += fabsf(a[base + c]);
  float r = blockReduceSum(s, red);
  if (tid == 0) atomicMaxPos(s1, r);
}
__global__ __launch_bounds__(128) void colmax_kernel(const float* __restrict__ a, float* s2)
{
  __shared__ float red[128];
  int h = blockIdx.x >> 9, j = blockIdx.x & 511, tid = threadIdx.x;
  long long base = (long long)h * MLM * MLM + j;
  float s = 0.0f;
  for (int i = tid; i < MLM; i += 128) s += fabsf(a[base + (long long)i * MLM]);
  float r = blockReduceSum(s, red);
  if (tid == 0) atomicMaxPos(s2, r);
}
__global__ void scaleinv_kernel(const float* s1, const float* s2, float* sc) { sc[0] = 1.0f / (s1[0] * s2[0]); }

__global__ __launch_bounds__(256) void ztrans_kernel(
    const float* __restrict__ a, const float* __restrict__ sc, bf16_t* __restrict__ z)
{
  long long idx = (long long)blockIdx.x * 256 + threadIdx.x; // 8*512*512
  int h = (int)(idx >> 18), r = (int)((idx >> 9) & 511), c = (int)(idx & 511);
  z[(long long)h * MLM * MLM + (long long)c * MLM + r] = f2bf(a[idx] * sc[0]);
}

// out = bf16(diag*I - in); optionally copy16 = bf16(in)
__global__ __launch_bounds__(256) void eps_kernel(
    const float* __restrict__ in, bf16_t* __restrict__ out,
    bf16_t* __restrict__ copy16, float diag)
{
  long long idx = (long long)blockIdx.x * 256 + threadIdx.x; // 8*512*512
  int i = (int)((idx >> 9) & 511), j = (int)(idx & 511);
  float v = in[idx];
  out[idx] = f2bf(((i == j) ? diag : 0.0f) - v);
  if (copy16) copy16[idx] = f2bf(v);
}

// ---------------- attn f32 + 33-tap depthwise conv(V) -> bf16 [np][1024] ----------------
__global__ __launch_bounds__(256) void convcombine_kernel(
    const float* __restrict__ attnF, const bf16_t* __restrict__ V,
    const float* __restrict__ rw, bf16_t* __restrict__ out)
{
  long long idx = (long long)blockIdx.x * 256 + threadIdx.x; // NP*DM
  int s = (int)(idx >> 10), c = (int)(idx & 1023);
  int h = c >> 7, dd = c & 127;
  long long vb = (long long)h * NP * DH + dd;
  float acc = attnF[(long long)h * NP * DH + (long long)s * DH + dd];
  #pragma unroll
  for (int t = 0; t < 33; ++t) {
    int ss = s + t - 16;
    if (ss >= 0 && ss < NP) acc += bf2f(V[vb + (long long)ss * DH]) * rw[h * 33 + t];
  }
  out[idx] = f2bf(acc);
}

// ---------------- PPEG ----------------
__global__ __launch_bounds__(256) void ppeg_head_kernel(
    const float* __restrict__ Y, float* __restrict__ X)
{
  int idx = blockIdx.x * 256 + threadIdx.x; // (PADR+1)*DM
  int row = idx >> 10;
  X[idx] = (row < PADR) ? 0.0f : Y[idx];
}

__global__ __launch_bounds__(256) void ppeg_conv_kernel(
    const float* __restrict__ Y, float* __restrict__ X,
    const float* __restrict__ w7, const float* __restrict__ b7,
    const float* __restrict__ w5, const float* __restrict__ b5,
    const float* __restrict__ w3, const float* __restrict__ b3)
{
  int idx = blockIdx.x * 256 + threadIdx.x; // DM * 10000
  int c = idx / (GH * GH), r = idx % (GH * GH);
  int i = r / GH, j = r % GH;
  const float* base = Y + (long long)(PADR + 1) * DM + c;
  float acc = base[(long long)r * DM] + b7[c] + b5[c] + b3[c];
  #pragma unroll
  for (int a = 0; a < 7; ++a) {
    int ii = i + a - 3; if (ii < 0 || ii >= GH) continue;
    #pragma unroll
    for (int bb = 0; bb < 7; ++bb) {
      int jj = j + bb - 3; if (jj < 0 || jj >= GH) continue;
      acc += base[(long long)(ii * GH + jj) * DM] * w7[c * 49 + a * 7 + bb];
    }
  }
  #pragma unroll
  for (int a = 0; a < 5; ++a) {
    int ii = i + a - 2; if (ii < 0 || ii >= GH) continue;
    #pragma unroll
    for (int bb = 0; bb < 5; ++bb) {
      int jj = j + bb - 2; if (jj < 0 || jj >= GH) continue;
      acc += base[(long long)(ii * GH + jj) * DM] * w5[c * 25 + a * 5 + bb];
    }
  }
  #pragma unroll
  for (int a = 0; a < 3; ++a) {
    int ii = i + a - 1; if (ii < 0 || ii >= GH) continue;
    #pragma unroll
    for (int bb = 0; bb < 3; ++bb) {
      int jj = j + bb - 1; if (jj < 0 || jj >= GH) continue;
      acc += base[(long long)(ii * GH + jj) * DM] * w3[c * 9 + a * 3 + bb];
    }
  }
  X[(long long)(PADR + 1 + r) * DM + c] = acc;
}

// ---------------- final: LN on CLS row + 1024x2 classifier ----------------
__global__ __launch_bounds__(256) void final_kernel(
    const float* __restrict__ xrow, const float* __restrict__ g, const float* __restrict__ b,
    const float* __restrict__ w, const float* __restrict__ bias, float* __restrict__ out)
{
  __shared__ float red[256];
  int tid = threadIdx.x;
  float s = 0.0f;
  for (int c = tid; c < DM; c += 256) s += xrow[c];
  float mu = blockReduceSum(s, red) * (1.0f / DM);
  float v = 0.0f;
  for (int c = tid; c < DM; c += 256) { float d = xrow[c] - mu; v += d * d; }
  float var = blockReduceSum(v, red) * (1.0f / DM);
  float rs = rsqrtf(var + 1e-5f);
  float a0 = 0.0f, a1 = 0.0f;
  for (int c = tid; c < DM; c += 256) {
    float xv = (xrow[c] - mu) * rs * g[c] + b[c];
    a0 += xv * w[c * 2]; a1 += xv * w[c * 2 + 1];
  }
  a0 = blockReduceSum(a0, red);
  a1 = blockReduceSum(a1, red);
  if (tid == 0) { out[0] = a0 + bias[0]; out[1] = a1 + bias[1]; }
}

// ---------------- f32 -> bf16 convert ----------------
__global__ __launch_bounds__(256) void cvt_kernel(const float* __restrict__ in, bf16_t* __restrict__ out, int n)
{
  int idx = blockIdx.x * 256 + threadIdx.x;
  if (idx < n) out[idx] = f2bf(in[idx]);
}

// ---------------- host-side epilogue builders ----------------
static EpArgs epF32(float* o, int ldo, long long so, float alpha = 1.0f) {
  EpArgs e{}; e.mode = 0; e.alpha = alpha; e.outF = o; e.ldo = ldo; e.strideO = so; return e;
}
static EpArgs epBF16(bf16_t* o, int ldo, long long so, float alpha = 1.0f) {
  EpArgs e{}; e.mode = 1; e.alpha = alpha; e.outB = o; e.ldo = ldo; e.strideO = so; return e;
}
static EpArgs epQKV(bf16_t* q, bf16_t* k, bf16_t* v, float qscale) {
  EpArgs e{}; e.mode = 2; e.alpha = 1.0f; e.qp = q; e.kp = k; e.vp = v;
  e.qscale = qscale; e.headStride = (long long)NP * DH; return e;
}
static EpArgs epBiasRes(float* o, int ldo, const float* bias, const float* resid, int zeroBelow) {
  EpArgs e{}; e.mode = 3; e.alpha = 1.0f; e.outF = o; e.ldo = ldo;
  e.bias = bias; e.resid = resid; e.zeroBelow = zeroBelow; return e;
}

static void gemm(hipStream_t st, const bf16_t* A, const bf16_t* B, int M, int N, int K,
                 int lda, int ldb, long long sA, long long sB, int batch, int transB, EpArgs ep) {
  dim3 g(N / BN, M / BM, batch), b(256);
  gemm_bf16_kernel<<<g, b, GEMM_SMEM, st>>>(A, B, M, N, K, lda, ldb, sA, sB, transB, ep);
}

// ---------------- orchestration ----------------
extern "C" void kernel_launch(void* const* d_in, const int* in_sizes, int n_in,
                              void* d_out, int out_size, void* d_ws, size_t ws_size,
                              hipStream_t stream)
{
  const float* h_    = (const float*)d_in[0];
  const float* wgt   = (const float*)d_in[1];
  const float* cls   = (const float*)d_in[2];
  const float* l1_ng = (const float*)d_in[3];
  const float* l1_nb = (const float*)d_in[4];
  const float* l1_qkv= (const float*)d_in[5];
  const float* l1_ow = (const float*)d_in[6];
  const float* l1_ob = (const float*)d_in[7];
  const float* l1_rw = (const float*)d_in[8];
  const float* l2_ng = (const float*)d_in[9];
  const float* l2_nb = (const float*)d_in[10];
  const float* l2_qkv= (const float*)d_in[11];
  const float* l2_ow = (const float*)d_in[12];
  const float* l2_ob = (const float*)d_in[13];
  const float* l2_rw = (const float*)d_in[14];
  const float* p7w = (const float*)d_in[15];
  const float* p7b = (const float*)d_in[16];
  const float* p5w = (const float*)d_in[17];
  const float* p5b = (const float*)d_in[18];
  const float* p3w = (const float*)d_in[19];
  const float* p3b = (const float*)d_in[20];
  const float* n_g = (const float*)d_in[21];
  const float* n_b = (const float*)d_in[22];
  const float* fc2w= (const float*)d_in[23];
  const float* fc2b= (const float*)d_in[24];

  char* ws = (char*)d_ws;
  size_t off = 0;
  auto alloc = [&](size_t bytes) -> void* {
    void* p = ws + off;
    off += (bytes + 255) & ~(size_t)255;
    return p;
  };

  const long long NPD  = (long long)NP * DM;         // 10,485,760
  const long long HM   = (long long)HEADS * MLM;     // 4096
  const long long MM   = (long long)MLM * MLM;       // 262,144
  const long long HMM  = HEADS * MM;                 // 2,097,152
  const long long HNPD = (long long)HEADS * NP * DH; // = NPD
  const long long AFN  = (long long)HEADS * MLM * NP; // 41,943,040

  float*  X     = (float*) alloc(NPD * 4);
  float*  Y     = (float*) alloc(NPD * 4);
  bf16_t* XN16  = (bf16_t*)alloc(NPD * 2);     // also aliased as ATTN16 (disjoint lifetimes)
  bf16_t* Q16   = (bf16_t*)alloc(HNPD * 2);
  bf16_t* K16   = (bf16_t*)alloc(HNPD * 2);
  bf16_t* V16   = (bf16_t*)alloc(HNPD * 2);
  bf16_t* QL16  = (bf16_t*)alloc(HM * DH * 2);
  bf16_t* KL16  = (bf16_t*)alloc(HM * DH * 2);
  float*  A2F   = (float*) alloc(HMM * 4);
  bf16_t* A2_16 = (bf16_t*)alloc(HMM * 2);
  bf16_t* Z16   = (bf16_t*)alloc(HMM * 2);
  bf16_t* Z2_16 = (bf16_t*)alloc(HMM * 2);
  bf16_t* XZ16  = (bf16_t*)alloc(HMM * 2);
  bf16_t* W16   = (bf16_t*)alloc(HMM * 2);
  float*  P1F   = (float*) alloc(HMM * 4);
  float*  AF    = (float*) alloc(AFN * 4);     // a3/a1 logits f32; also aliased as ATTNF
  bf16_t* A16   = (bf16_t*)alloc(AFN * 2);     // a3/a1 probabilities bf16
  bf16_t* T1_16 = (bf16_t*)alloc(HM * DH * 2);
  bf16_t* T2_16 = (bf16_t*)alloc(HM * DH * 2);
  bf16_t* qw16a = (bf16_t*)alloc((size_t)DM * 3 * DM * 2);
  bf16_t* qw16b = (bf16_t*)alloc((size_t)DM * 3 * DM * 2);
  bf16_t* ow16a = (bf16_t*)alloc((size_t)DM * DM * 2);
  bf16_t* ow16b = (bf16_t*)alloc((size_t)DM * DM * 2);
  float*  scal  = (float*) alloc(64 * 4);      // s1, s2, scaleInv

  float* ATTNF = AF;          // alias: free once a1 softmax has produced A16
  bf16_t* ATTN16 = XN16;      // alias: free once qkv GEMM consumed XN16

  const float qscale = 0.08838834764831845f;   // 1/sqrt(128)

  // weight conversions (f32 -> bf16)
  cvt_kernel<<<(DM * 3 * DM) / 256, 256, 0, stream>>>(l1_qkv, qw16a, DM * 3 * DM);
  cvt_kernel<<<(DM * 3 * DM) / 256, 256, 0, stream>>>(l2_qkv, qw16b, DM * 3 * DM);
  cvt_kernel<<<(DM * DM) / 256, 256, 0, stream>>>(l1_ow, ow16a, DM * DM);
  cvt_kernel<<<(DM * DM) / 256, 256, 0, stream>>>(l2_ow, ow16b, DM * DM);

  // build input sequence
  build_kernel<<<(int)(NPD / 256), 256, 0, stream>>>(h_, wgt, cls, X);

  auto layer = [&](const float* ng, const float* nb, const bf16_t* qkv16, const bf16_t* ow16,
                   const float* ob, const float* rw, float* Xi, float* Yo) {
    // LayerNorm (in place) + bf16 copy
    ln_kernel<<<NP, 256, 0, stream>>>(Xi, XN16, ng, nb);
    // QKV projection: [NP,1024] x [1024,3072] -> per-head q,k,v (q scaled)
    gemm(stream, XN16, qkv16, NP, 3 * DM, DM, DM, 3 * DM, 0, 0, 1, 0,
         epQKV(Q16, K16, V16, qscale));
    // landmark means
    landmark_kernel<<<HEADS * MLM, DH, 0, stream>>>(Q16, K16, QL16, KL16);
    // a2 = softmax(q_l @ k_l^T)
    gemm(stream, QL16, KL16, MLM, MLM, DH, DH, DH, (long long)MLM * DH,
         (long long)MLM * DH, HEADS, 1, epF32(A2F, MLM, MM));
    softmax_kernel<<<HEADS * MLM, 256, 0, stream>>>(A2F, A2F, A2_16, MLM);
    // Moore-Penrose pinv scale
    init_scal_kernel<<<1, 1, 0, stream>>>(scal, scal + 1);
    rowmax_kernel<<<HEADS * MLM, 128, 0, stream>>>(A2F, scal);
    colmax_kernel<<<HEADS * MLM, 128, 0, stream>>>(A2F, scal + 1);
    scaleinv_kernel<<<1, 1, 0, stream>>>(scal, scal + 1, scal + 2);
    ztrans_kernel<<<(int)(HMM / 256), 256, 0, stream>>>(A2F, scal + 2, Z16);
    // 6 Newton-Schulz iterations (bf16 WMMA, f32 accumulate)
    bf16_t* zc = Z16; bf16_t* zn = Z2_16;
    for (int it = 0; it < 6; ++it) {
      gemm(stream, A2_16, zc, MLM, MLM, MLM, MLM, MLM, MM, MM, HEADS, 0, epF32(P1F, MLM, MM));
      eps_kernel<<<(int)(HMM / 256), 256, 0, stream>>>(P1F, W16, XZ16, 7.0f);
      gemm(stream, XZ16, W16, MLM, MLM, MLM, MLM, MLM, MM, MM, HEADS, 0, epF32(P1F, MLM, MM));
      eps_kernel<<<(int)(HMM / 256), 256, 0, stream>>>(P1F, W16, (bf16_t*)nullptr, 15.0f);
      gemm(stream, XZ16, W16, MLM, MLM, MLM, MLM, MLM, MM, MM, HEADS, 0, epF32(P1F, MLM, MM));
      eps_kernel<<<(int)(HMM / 256), 256, 0, stream>>>(P1F, W16, (bf16_t*)nullptr, 13.0f);
      gemm(stream, zc, W16, MLM, MLM, MLM, MLM, MLM, MM, MM, HEADS, 0,
           epBF16(zn, MLM, MM, 0.25f));
      bf16_t* t = zc; zc = zn; zn = t;
    }
    // a3 = softmax(q_l @ k^T)  [h,512,10240]
    gemm(stream, QL16, K16, MLM, NP, DH, DH, DH, (long long)MLM * DH, (long long)NP * DH,
         HEADS, 1, epF32(AF, NP, (long long)MLM * NP));
    softmax_kernel<<<HEADS * MLM, 256, 0, stream>>>(AF, (float*)nullptr, A16, NP);
    // t1 = a3 @ v  [h,512,128]
    gemm(stream, A16, V16, MLM, DH, NP, NP, DH, (long long)MLM * NP, (long long)NP * DH,
         HEADS, 0, epBF16(T1_16, DH, (long long)MLM * DH));
    // t2 = a2_inv @ t1
    gemm(stream, zc, T1_16, MLM, DH, MLM, MLM, DH, MM, (long long)MLM * DH, HEADS, 0,
         epBF16(T2_16, DH, (long long)MLM * DH));
    // a1 = softmax(q @ k_l^T)  [h,10240,512]
    gemm(stream, Q16, KL16, NP, MLM, DH, DH, DH, (long long)NP * DH, (long long)MLM * DH,
         HEADS, 1, epF32(AF, MLM, (long long)NP * MLM));
    softmax_kernel<<<HEADS * NP, 256, 0, stream>>>(AF, (float*)nullptr, A16, MLM);
    // attn = a1 @ t2  [h,10240,128]
    gemm(stream, A16, T2_16, NP, DH, MLM, MLM, DH, (long long)NP * MLM, (long long)MLM * DH,
         HEADS, 0, epF32(ATTNF, DH, (long long)NP * DH));
    // + depthwise 33-tap conv residual, merge heads, -> bf16 [NP,1024]
    convcombine_kernel<<<(int)(NPD / 256), 256, 0, stream>>>(ATTNF, V16, rw, ATTN16);
    // out projection + bias + residual(normalized x); zero pad rows
    gemm(stream, ATTN16, ow16, NP, DM, DM, DM, DM, 0, 0, 1, 0,
         epBiasRes(Yo, DM, ob, Xi, PADR));
  };

  // layer 1: X -> Y
  layer(l1_ng, l1_nb, qw16a, ow16a, l1_ob, l1_rw, X, Y);

  // PPEG: Y -> X
  ppeg_head_kernel<<<((PADR + 1) * DM) / 256, 256, 0, stream>>>(Y, X);
  ppeg_conv_kernel<<<(DM * GH * GH) / 256, 256, 0, stream>>>(Y, X, p7w, p7b, p5w, p5b, p3w, p3b);

  // layer 2: X -> Y
  layer(l2_ng, l2_nb, qw16b, ow16b, l2_ob, l2_rw, X, Y);

  // final LN on CLS token + classifier
  final_kernel<<<1, 256, 0, stream>>>(Y + (long long)PADR * DM, n_g, n_b, fc2w, fc2b,
                                      (float*)d_out);

  (void)in_sizes; (void)n_in; (void)out_size; (void)ws_size;
}
